// TempestBlock_37177236914307
// MI455X (gfx1250) — matrix operand order repaired and checked by reading, not compiled, not run on hardware
//
#include <hip/hip_runtime.h>
#include <hip/hip_bf16.h>

// ---------------------------------------------------------------------------
// Types for CDNA5 WMMA
// ---------------------------------------------------------------------------
typedef __bf16 bf16_t;
typedef __attribute__((ext_vector_type(16))) __bf16 v16bf;
typedef __attribute__((ext_vector_type(8)))  __bf16 v8bf;
typedef __attribute__((ext_vector_type(8)))  float  v8f;

__device__ __forceinline__ bf16_t f2bf(float f) {
  union { float f; unsigned u; } in; in.f = f;
  unsigned u = in.u;
  unsigned r = u + 0x7FFFu + ((u >> 16) & 1u);   // round-to-nearest-even
  unsigned short h = (unsigned short)(r >> 16);
  bf16_t out;
  __builtin_memcpy(&out, &h, 2);
  return out;
}

__device__ __forceinline__ float dsigmoid(float x) {
  return 1.0f / (1.0f + __expf(-x));
}

// ---------------------------------------------------------------------------
// Tiled bf16 GEMM with v_wmma_f32_16x16x32_bf16.
//   C[M,N] (f32) = A[M,K] (bf16, row-major) * B[K,N] (bf16, row-major)
//   + optional addend[M,N] (f32)
// Block: 256 threads = 8 waves arranged 4x2. Tile: BM=128, BN=128, BK=32.
// Wave tile: 32x64 => 2x4 WMMA 16x16 accumulators.
// Requires: M%128==0, N%128==0, K%32==0 (true for all shapes here).
// ---------------------------------------------------------------------------
#define BM 128
#define BN 128
#define BK 32

__global__ __launch_bounds__(256)
void gemm_bf16_wmma(const bf16_t* __restrict__ A, const bf16_t* __restrict__ B,
                    float* __restrict__ C, const float* __restrict__ addend,
                    int M, int N, int K)
{
  __shared__ bf16_t As[2][BM][BK];   // [m][k] row-major
  __shared__ bf16_t Bs[2][BN][BK];   // [n][k] transposed

  const int tid  = threadIdx.x;
  const int lane = tid & 31;
  const int wv   = tid >> 5;      // 0..7
  const int wm   = wv >> 1;       // 0..3 -> 32-row strip
  const int wn   = wv & 1;        // 0..1 -> 64-col strip
  const int fr   = lane & 15;     // lane row/col within 16
  const int half = lane >> 4;     // 0 or 1 (K-half / M-half selector)
  const int m0 = blockIdx.y * BM;
  const int n0 = blockIdx.x * BN;

  v8f acc[2][4];
#pragma unroll
  for (int i = 0; i < 2; ++i)
#pragma unroll
    for (int j = 0; j < 4; ++j)
#pragma unroll
      for (int r = 0; r < 8; ++r) acc[i][j][r] = 0.0f;

  auto load_tiles = [&](int kblk, int buf) {
    // A tile: 128x32 = 512 chunks of 8 bf16 (16B) -> 2 chunks/thread, coalesced
    for (int c = tid; c < 512; c += 256) {
      const int r  = c >> 2;
      const int cc = (c & 3) << 3;
      const bf16_t* src = A + (size_t)(m0 + r) * K + kblk + cc;
      *reinterpret_cast<v8bf*>(&As[buf][r][cc]) =
          *reinterpret_cast<const v8bf*>(src);
    }
    // B tile: 32x128, read coalesced rows, write transposed into Bs[n][k]
    for (int c = tid; c < 512; c += 256) {
      const int kr = c >> 4;
      const int nc = (c & 15) << 3;
      const bf16_t* src = B + (size_t)(kblk + kr) * N + n0 + nc;
      v8bf v = *reinterpret_cast<const v8bf*>(src);
#pragma unroll
      for (int j = 0; j < 8; ++j) Bs[buf][nc + j][kr] = v[j];
    }
  };

  auto compute = [&](int buf) {
    v16bf afrag[2];
    v16bf bfrag[4];
    // A 16x32 bf16 layout: lanes 0-15 -> M=fr, K={0..7,16..23};
    //                      lanes 16-31 -> M=fr, K={8..15,24..31}
#pragma unroll
    for (int mt = 0; mt < 2; ++mt) {
      const bf16_t* ap = &As[buf][wm * 32 + mt * 16 + fr][half * 8];
      v8bf lo = *reinterpret_cast<const v8bf*>(ap);
      v8bf hi = *reinterpret_cast<const v8bf*>(ap + 16);
      afrag[mt] = __builtin_shufflevector(lo, hi, 0,1,2,3,4,5,6,7,8,9,10,11,12,13,14,15);
    }
    // B 32x16 bf16 layout: lanes 0-15 -> N=fr, K={0..7,16..23}; lanes 16-31 -> N=fr, K={8..15,24..31}
#pragma unroll
    for (int nt = 0; nt < 4; ++nt) {
      const bf16_t* bp = &Bs[buf][wn * 64 + nt * 16 + fr][half * 8];
      v8bf lo = *reinterpret_cast<const v8bf*>(bp);
      v8bf hi = *reinterpret_cast<const v8bf*>(bp + 16);
      bfrag[nt] = __builtin_shufflevector(lo, hi, 0,1,2,3,4,5,6,7,8,9,10,11,12,13,14,15);
    }
#pragma unroll
    for (int mt = 0; mt < 2; ++mt)
#pragma unroll
      for (int nt = 0; nt < 4; ++nt)
        acc[mt][nt] = __builtin_amdgcn_wmma_f32_16x16x32_bf16(
            false, afrag[mt], false, bfrag[nt],
            (short)0, acc[mt][nt], false, false);
  };

  const int nk = K >> 5;
  load_tiles(0, 0);
  __syncthreads();
  for (int kt = 0; kt < nk; ++kt) {
    const int buf = kt & 1;
    if (kt + 2 < nk) {   // prefetch 2 tiles ahead -> global_prefetch_b8
      __builtin_prefetch(A + (size_t)(m0 + (tid >> 1)) * K + ((kt + 2) << 5), 0, 1);
      __builtin_prefetch(B + (size_t)(((kt + 2) << 5) + (tid >> 3)) * N + n0, 0, 1);
    }
    if (kt + 1 < nk) load_tiles((kt + 1) << 5, buf ^ 1);
    compute(buf);
    __syncthreads();
  }

  // Epilogue: C 16x16 f32 layout: VGPR r, lanes 0-15 -> (M=r, N=fr);
  //                               lanes 16-31 -> (M=8+r, N=fr)
#pragma unroll
  for (int mt = 0; mt < 2; ++mt)
#pragma unroll
    for (int nt = 0; nt < 4; ++nt) {
      const int row = m0 + wm * 32 + mt * 16 + half * 8;
      const int col = n0 + wn * 64 + nt * 16 + fr;
#pragma unroll
      for (int r = 0; r < 8; ++r) {
        const size_t idx = (size_t)(row + r) * N + col;
        float v = acc[mt][nt][r];
        if (addend) v += addend[idx];
        C[idx] = v;
      }
    }
}

// ---------------------------------------------------------------------------
// Weight cast/pack: dst[r][dstColOff + c] (bf16, ld=dstLd) = src[r][c] (f32)
// ---------------------------------------------------------------------------
__global__ void cast_pack_kernel(const float* __restrict__ src, bf16_t* __restrict__ dst,
                                 int rows, int cols, int dstLd, int dstColOff)
{
  const long i = (long)blockIdx.x * blockDim.x + threadIdx.x;
  const long total = (long)rows * cols;
  if (i >= total) return;
  const int r = (int)(i / cols);
  const int c = (int)(i % cols);
  dst[(size_t)r * dstLd + dstColOff + c] = f2bf(src[i]);
}

// ---------------------------------------------------------------------------
// RMSNorm + cast to bf16. One block (256 thr) per row of D elements.
// ---------------------------------------------------------------------------
__global__ void rmsnorm_cast_kernel(const float* __restrict__ x, const float* __restrict__ w,
                                    bf16_t* __restrict__ out, int D)
{
  __shared__ float red[256];
  const int row = blockIdx.x;
  const int tid = threadIdx.x;
  const float* xr = x + (size_t)row * D;
  float ss = 0.0f;
  for (int i = tid; i < D; i += 256) { float v = xr[i]; ss += v * v; }
  red[tid] = ss;
  __syncthreads();
  for (int s = 128; s > 0; s >>= 1) {
    if (tid < s) red[tid] += red[tid + s];
    __syncthreads();
  }
  const float rnorm = rsqrtf(red[0] / (float)D + 1e-6f);
  for (int i = tid; i < D; i += 256)
    out[(size_t)row * D + i] = f2bf(xr[i] * rnorm * w[i]);
}

// ---------------------------------------------------------------------------
// GatedConv pieces
// ---------------------------------------------------------------------------
__global__ void gate_mul_kernel(const float* __restrict__ bch, float* __restrict__ y,
                                int M, int DC)
{
  const long i = (long)blockIdx.x * blockDim.x + threadIdx.x;
  if (i >= (long)M * DC) return;
  const long row = i / DC; const int c = (int)(i % DC);
  y[i] = bch[(size_t)row * (3 * DC) + c] * bch[(size_t)row * (3 * DC) + 2 * DC + c];
}

// z[t] = w0*y[t-2] + w1*y[t-1] + w2*y[t] + b;  out = c_part * z -> bf16 into mix[:,0:DC]
__global__ void dwconv_kernel(const float* __restrict__ y, const float* __restrict__ bch,
                              const float* __restrict__ w, const float* __restrict__ bias,
                              bf16_t* __restrict__ mix, int M, int T, int DC, int DMIX)
{
  const long i = (long)blockIdx.x * blockDim.x + threadIdx.x;
  if (i >= (long)M * DC) return;
  const int ch = (int)(i % DC);
  const long row = i / DC;             // global token index
  const int t = (int)(row % T);        // time within a batch sequence
  const float y0 = y[i];
  const float y1 = (t >= 1) ? y[i - DC] : 0.0f;
  const float y2 = (t >= 2) ? y[i - 2 * DC] : 0.0f;
  const float z = w[ch * 3 + 0] * y2 + w[ch * 3 + 1] * y1 + w[ch * 3 + 2] * y0 + bias[ch];
  const float cp = bch[(size_t)row * (3 * DC) + DC + ch];
  mix[(size_t)row * DMIX + ch] = f2bf(cp * z);
}

// ---------------------------------------------------------------------------
// Griffin gates:  a = sigmoid(aiv[:,0:DR]+db), i = sigmoid(aiv[:,DR:2DR]),
//                 s = sqrt(max(1-a^2,1e-8)) * i * aiv[:,2DR:3DR]
// ---------------------------------------------------------------------------
__global__ void griffin_gates_kernel(const float* __restrict__ aiv, const float* __restrict__ db,
                                     float* __restrict__ a, float* __restrict__ s,
                                     int M, int DR)
{
  const long i = (long)blockIdx.x * blockDim.x + threadIdx.x;
  if (i >= (long)M * DR) return;
  const long row = i / DR; const int c = (int)(i % DR);
  const float* r = aiv + (size_t)row * (3 * DR);
  const float av = dsigmoid(r[c] + db[c]);
  const float iv = dsigmoid(r[DR + c]);
  const float vv = r[2 * DR + c];
  a[i] = av;
  s[i] = sqrtf(fmaxf(1.0f - av * av, 1e-8f)) * iv * vv;
}

// ---------------------------------------------------------------------------
// Chunked linear recurrence h[t] = a[t]*h[t-1] + s[t].  CH=64, NC=T/64.
// ---------------------------------------------------------------------------
__global__ void scan_phase1(const float* __restrict__ a, const float* __restrict__ s,
                            float* __restrict__ pro, float* __restrict__ end,
                            int Bn, int T, int DR)
{
  const int NC = T / 64;
  const long i = (long)blockIdx.x * blockDim.x + threadIdx.x;
  if (i >= (long)Bn * NC * DR) return;
  const int d = (int)(i % DR);
  const int ch = (int)((i / DR) % NC);
  const int b = (int)(i / ((long)DR * NC));
  const size_t base = ((size_t)b * T + (size_t)ch * 64) * DR + d;
  float h = 0.0f, p = 1.0f;
  for (int t = 0; t < 64; ++t) {
    const float av = a[base + (size_t)t * DR];
    h = av * h + s[base + (size_t)t * DR];
    p *= av;
  }
  pro[i] = p; end[i] = h;
}

__global__ void scan_phase2(const float* __restrict__ pro, const float* __restrict__ end,
                            float* __restrict__ carry, int Bn, int T, int DR)
{
  const int NC = T / 64;
  const long i = (long)blockIdx.x * blockDim.x + threadIdx.x;
  if (i >= (long)Bn * DR) return;
  const int d = (int)(i % DR);
  const int b = (int)(i / DR);
  float h = 0.0f;
  for (int ch = 0; ch < NC; ++ch) {
    const size_t o = ((size_t)b * NC + ch) * DR + d;
    carry[o] = h;
    h = pro[o] * h + end[o];
  }
}

__global__ void scan_phase3(const float* __restrict__ a, const float* __restrict__ s,
                            const float* __restrict__ carry, bf16_t* __restrict__ mix,
                            int Bn, int T, int DR, int DC, int DMIX)
{
  const int NC = T / 64;
  const long i = (long)blockIdx.x * blockDim.x + threadIdx.x;
  if (i >= (long)Bn * NC * DR) return;
  const int d = (int)(i % DR);
  const int ch = (int)((i / DR) % NC);
  const int b = (int)(i / ((long)DR * NC));
  const size_t base = ((size_t)b * T + (size_t)ch * 64) * DR + d;
  float h = carry[i];
  for (int t = 0; t < 64; ++t) {
    const size_t o = base + (size_t)t * DR;
    h = a[o] * h + s[o];
    mix[((size_t)b * T + (size_t)ch * 64 + t) * DMIX + DC + d] = f2bf(h);
  }
}

// ---------------------------------------------------------------------------
// Momentum residual: vel_new = sigmoid(log_beta)*vel + mixer; h2 = x + vel_new
// ---------------------------------------------------------------------------
__global__ void momentum_kernel(const float* __restrict__ x, const float* __restrict__ vel,
                                const float* __restrict__ mixer, const float* __restrict__ log_beta,
                                float* __restrict__ h2, float* __restrict__ velOut, long n)
{
  const long i = (long)blockIdx.x * blockDim.x + threadIdx.x;
  if (i >= n) return;
  const float beta = dsigmoid(log_beta[0]);
  const float vn = beta * vel[i] + mixer[i];
  velOut[i] = vn;
  h2[i] = x[i] + vn;
}

// ---------------------------------------------------------------------------
// SwiGLU: t = silu(gate) * up  (gate=gu[:,0:F], up=gu[:,F:2F]) -> bf16
// ---------------------------------------------------------------------------
__global__ void swiglu_kernel(const float* __restrict__ gu, bf16_t* __restrict__ t,
                              int M, int F)
{
  const long i = (long)blockIdx.x * blockDim.x + threadIdx.x;
  if (i >= (long)M * F) return;
  const long row = i / F; const int c = (int)(i % F);
  const float g = gu[(size_t)row * (2 * F) + c];
  const float u = gu[(size_t)row * (2 * F) + F + c];
  t[i] = f2bf(g * dsigmoid(g) * u);
}

// ---------------------------------------------------------------------------
// Host launcher
// ---------------------------------------------------------------------------
extern "C" void kernel_launch(void* const* d_in, const int* in_sizes, int n_in,
                              void* d_out, int out_size, void* d_ws, size_t ws_size,
                              hipStream_t stream)
{
  const int Bn = 4, T = 4096, D = 1024, DC = 640, DR = 384, FFN = 2560;
  const int M = Bn * T;           // 16384 tokens
  const int NC = T / 64;          // 64 chunks

  const float* x         = (const float*)d_in[0];
  const float* velocity  = (const float*)d_in[1];
  const float* pre_w     = (const float*)d_in[2];
  const float* conv_proj = (const float*)d_in[3];
  const float* conv_w    = (const float*)d_in[4];
  const float* conv_b    = (const float*)d_in[5];
  const float* w_a       = (const float*)d_in[6];
  const float* w_i       = (const float*)d_in[7];
  const float* w_v       = (const float*)d_in[8];
  const float* dbias     = (const float*)d_in[9];
  const float* out_proj  = (const float*)d_in[10];
  const float* log_beta  = (const float*)d_in[11];
  const float* ffn_w     = (const float*)d_in[12];
  const float* w_gu      = (const float*)d_in[13];
  const float* w_down    = (const float*)d_in[14];

  float* out_main = (float*)d_out;                       // [B,T,D]
  float* out_vel  = (float*)d_out + (size_t)M * D;       // [B,T,D]

  // Bump allocator over workspace
  char* wsb = (char*)d_ws;
  size_t off = 0;
  auto alloc = [&](size_t bytes) -> void* {
    off = (off + 255) & ~(size_t)255;
    void* p = wsb + off;
    off += bytes;
    return p;
  };

  // bf16 weights (persistent)
  bf16_t* wc_conv = (bf16_t*)alloc((size_t)D * 3 * DC * 2);
  bf16_t* wc_aiv  = (bf16_t*)alloc((size_t)D * 3 * DR * 2);
  bf16_t* wc_out  = (bf16_t*)alloc((size_t)D * D * 2);
  bf16_t* wc_gu   = (bf16_t*)alloc((size_t)D * 2 * FFN * 2);
  bf16_t* wc_down = (bf16_t*)alloc((size_t)FFN * D * 2);
  // activations
  bf16_t* xn   = (bf16_t*)alloc((size_t)M * D * 2);
  bf16_t* mix  = (bf16_t*)alloc((size_t)M * D * 2);      // [conv_out | griffin_out] bf16
  float*  h2   = (float*)alloc((size_t)M * D * 4);
  bf16_t* h2n  = (bf16_t*)alloc((size_t)M * D * 2);
  float*  abuf = (float*)alloc((size_t)M * DR * 4);
  float*  sbuf = (float*)alloc((size_t)M * DR * 4);
  float*  pro  = (float*)alloc((size_t)Bn * NC * DR * 4);
  float*  endb = (float*)alloc((size_t)Bn * NC * DR * 4);
  float*  carr = (float*)alloc((size_t)Bn * NC * DR * 4);
  float*  ybuf = (float*)alloc((size_t)M * DC * 4);
  bf16_t* tbuf = (bf16_t*)alloc((size_t)M * FFN * 2);
  // big fp32 scratch, reused serially: bch -> aiv -> mixer_out -> gate_up
  float* scratchA = (float*)alloc((size_t)M * 2 * FFN * 4);   // max N = 5120
  (void)ws_size; (void)n_in; (void)in_sizes; (void)out_size;

  auto gemm = [&](const bf16_t* A, const bf16_t* Bw, float* C, const float* add,
                  int m, int n, int k) {
    dim3 grid(n / BN, m / BM);
    gemm_bf16_wmma<<<grid, 256, 0, stream>>>(A, Bw, C, add, m, n, k);
  };
  auto ew = [&](long n) { return dim3((unsigned)((n + 255) / 256)); };

  // --- Weight casts / packing -------------------------------------------
  cast_pack_kernel<<<ew((long)D * 3 * DC), 256, 0, stream>>>(conv_proj, wc_conv, D, 3 * DC, 3 * DC, 0);
  cast_pack_kernel<<<ew((long)D * DR), 256, 0, stream>>>(w_a, wc_aiv, D, DR, 3 * DR, 0);
  cast_pack_kernel<<<ew((long)D * DR), 256, 0, stream>>>(w_i, wc_aiv, D, DR, 3 * DR, DR);
  cast_pack_kernel<<<ew((long)D * DR), 256, 0, stream>>>(w_v, wc_aiv, D, DR, 3 * DR, 2 * DR);
  cast_pack_kernel<<<ew((long)D * D), 256, 0, stream>>>(out_proj, wc_out, D, D, D, 0);
  cast_pack_kernel<<<ew((long)D * 2 * FFN), 256, 0, stream>>>(w_gu, wc_gu, D, 2 * FFN, 2 * FFN, 0);
  cast_pack_kernel<<<ew((long)FFN * D), 256, 0, stream>>>(w_down, wc_down, FFN, D, D, 0);

  // --- Pre-norm ----------------------------------------------------------
  rmsnorm_cast_kernel<<<M, 256, 0, stream>>>(x, pre_w, xn, D);

  // --- GatedConv ---------------------------------------------------------
  float* bch = scratchA;                                  // [M, 1920]
  gemm(xn, wc_conv, bch, nullptr, M, 3 * DC, D);
  gate_mul_kernel<<<ew((long)M * DC), 256, 0, stream>>>(bch, ybuf, M, DC);
  dwconv_kernel<<<ew((long)M * DC), 256, 0, stream>>>(ybuf, bch, conv_w, conv_b, mix, M, T, DC, D);

  // --- Griffin recurrence ------------------------------------------------
  float* aiv = scratchA;                                  // [M, 1152] (bch dead)
  gemm(xn, wc_aiv, aiv, nullptr, M, 3 * DR, D);
  griffin_gates_kernel<<<ew((long)M * DR), 256, 0, stream>>>(aiv, dbias, abuf, sbuf, M, DR);
  scan_phase1<<<ew((long)Bn * NC * DR), 256, 0, stream>>>(abuf, sbuf, pro, endb, Bn, T, DR);
  scan_phase2<<<ew((long)Bn * DR), 256, 0, stream>>>(pro, endb, carr, Bn, T, DR);
  scan_phase3<<<ew((long)Bn * NC * DR), 256, 0, stream>>>(abuf, sbuf, carr, mix, Bn, T, DR, DC, D);

  // --- Mixer out-proj + momentum residual --------------------------------
  float* mixer = scratchA;                                // [M, 1024] (aiv dead)
  gemm(mix, wc_out, mixer, nullptr, M, D, D);
  momentum_kernel<<<ew((long)M * D), 256, 0, stream>>>(x, velocity, mixer, log_beta,
                                                       h2, out_vel, (long)M * D);

  // --- FFN ---------------------------------------------------------------
  rmsnorm_cast_kernel<<<M, 256, 0, stream>>>(h2, ffn_w, h2n, D);
  float* gu = scratchA;                                   // [M, 5120] (mixer dead)
  gemm(h2n, wc_gu, gu, nullptr, M, 2 * FFN, D);
  swiglu_kernel<<<ew((long)M * FFN), 256, 0, stream>>>(gu, tbuf, M, FFN);
  gemm(tbuf, wc_down, out_main, h2, M, D, FFN);           // fused residual into d_out
}
